// MultiAttention_82738249990568
// MI455X (gfx1250) — matrix operand order repaired and checked by reading, not compiled, not run on hardware
//
#include <hip/hip_runtime.h>
#include <hip/hip_bf16.h>

#define BB 32
#define SS 4096
#define KK 512
#define HH 512
#define WAVES 4            // waves per scores-block (64 s-rows per block)
#define NTILES (HH / 16)   // 32 n-tiles
#define PANEL_H (KK * 16)  // f16 elements per B panel (8192 = 16KB)

typedef __attribute__((ext_vector_type(16))) _Float16 v16h;
typedef __attribute__((ext_vector_type(8)))  _Float16 v8h;
typedef __attribute__((ext_vector_type(4)))  _Float16 v4h;
typedef __attribute__((ext_vector_type(8)))  float    v8f;
typedef __attribute__((ext_vector_type(4)))  float    v4f;
typedef __attribute__((ext_vector_type(4)))  unsigned int v4u;
typedef __attribute__((ext_vector_type(8)))  int      v8i;
typedef __attribute__((ext_vector_type(4)))  int      v4i;

union F16Frag { v16h v; v8h h[2]; };

#if __has_builtin(__builtin_amdgcn_tensor_load_to_lds) && __has_builtin(__builtin_amdgcn_s_wait_tensorcnt)
#define USE_TDM 1
#else
#define USE_TDM 0
#endif

#if USE_TDM
// Issue one TDM op: contiguous 16KB (4096 dwords) global -> LDS.
// D# built per CDNA5 ISA (1-row tile, dword units).
// This toolchain exposes the 6-arg builtin:
//   (uint32x4 g0, int32x8 g1, int32x4 g2, int32x4 g3, int32x8 extra, i32 cpol)
__device__ __forceinline__ void tdm_load_panel(const _Float16* gsrc, _Float16* ldst) {
    const unsigned long long ga = (unsigned long long)(uintptr_t)gsrc;
    const unsigned int lds_off  = (unsigned int)(uintptr_t)ldst; // addr[31:0] = LDS byte addr
    v4u g0;
    g0.x = 1u;                                   // count=1, is_restore=0, gather=0
    g0.y = lds_off;                              // lds_addr (bytes)
    g0.z = (unsigned int)ga;                     // global_addr[31:0]
    g0.w = (unsigned int)(ga >> 32) | (2u << 30);// global_addr[56:32] | type=2
    v8i g1;
    g1[0] = 0x20000;                // workgroup_mask=0 | data_size=2 (4-byte)
    g1[1] = (int)(4096u << 16);     // tensor_dim0[15:0] (=4096 dwords) in bits 63:48
    g1[2] = 0x00010000;             // tensor_dim0[31:16]=0 | tensor_dim1=1
    g1[3] = (int)(4096u << 16);     // tile_dim0 = 4096 dwords (bits 127:112)
    g1[4] = 1;                      // tile_dim1=1, tile_dim2=0
    g1[5] = 4096;                   // tensor_dim0_stride lo
    g1[6] = 0;
    g1[7] = 0;
    v4i z4; z4[0] = 0; z4[1] = 0; z4[2] = 0; z4[3] = 0;
    v8i z8; z8[0] = 0; z8[1] = 0; z8[2] = 0; z8[3] = 0;
               z8[4] = 0; z8[5] = 0; z8[6] = 0; z8[7] = 0;
    __builtin_amdgcn_tensor_load_to_lds(g0, g1, z4, z4, z8, 0);
}
#endif

// ---------------------------------------------------------------------------
// Kernel 0: pack Wk (f32 row-major [K,H]) -> f16 in B-fragment order:
// per (head, nt): f16 idx = ((ks*32 + n + (kk&16))*16 + (kk&15)),  k = ks*32+kk
// ---------------------------------------------------------------------------
__global__ void pack_wk_kernel(const float* __restrict__ wka,
                               const float* __restrict__ wkb,
                               _Float16* __restrict__ wkp) {
    const int head = blockIdx.x;   // 2
    const int nt   = blockIdx.y;   // 32
    const float* Wk = head ? wkb : wka;
    _Float16* dst = wkp + ((size_t)head * NTILES + nt) * PANEL_H;
    for (int e = threadIdx.x; e < PANEL_H; e += 256) {
        const int k = e >> 4, n = e & 15;
        const int kk = k & 31, ks = k >> 5;
        const int dl = n + (kk & 16);
        dst[(ks * 32 + dl) * 16 + (kk & 15)] =
            (_Float16)Wk[(size_t)k * HH + nt * 16 + n];
    }
}

// ---------------------------------------------------------------------------
// Kernel 1: proj_q[head][b][h] = sum_i query[b,i] * Wq[i,h]   (tiny GEMM)
// ---------------------------------------------------------------------------
__global__ void pq_kernel(const float* __restrict__ query,
                          const float* __restrict__ Wq_a,
                          const float* __restrict__ Wq_b,
                          float* __restrict__ pq) {
    const int b = blockIdx.x, head = blockIdx.y, h = threadIdx.x;
    const float* Wq = head ? Wq_b : Wq_a;
    const float* q  = query + (size_t)b * HH;
    float acc = 0.f;
    for (int i = 0; i < HH; ++i) acc += q[i] * Wq[(size_t)i * HH + h];
    pq[((size_t)head * BB + b) * HH + h] = acc;
}

// ---------------------------------------------------------------------------
// Kernel 2: fused  scores[b,s] = v . tanh(proj_q[b] + values[b,s,:] @ Wk)
// 4 waves / block; A tiles (f16) in LDS; Wk panels streamed into LDS by the
// Tensor Data Mover, double-buffered against the WMMA k-loop.
// ---------------------------------------------------------------------------
__launch_bounds__(WAVES * 32)
__global__ void scores_kernel(const float* __restrict__ va,
                              const float* __restrict__ vb,
                              const _Float16* __restrict__ wkp,
                              const float* __restrict__ vveca,
                              const float* __restrict__ vvecb,
                              const float* __restrict__ pq,
                              float* __restrict__ scores) {
    const int head  = blockIdx.z;
    const int b     = blockIdx.y;
    const int sbase = blockIdx.x * (WAVES * 16);

    const float* V   = head ? vb    : va;
    const float* vv  = head ? vvecb : vveca;
    const float* pqh = pq + ((size_t)head * BB + b) * HH;
    const _Float16* panels = wkp + (size_t)head * NTILES * PANEL_H;

    __shared__ alignas(16) _Float16 ldsA[WAVES][16][KK + 8];
    __shared__ alignas(16) _Float16 ldsB[2][PANEL_H];

    const int tid  = threadIdx.x;
    const int wave = tid >> 5;
    const int lane = tid & 31;
    const int row  = lane & 15;   // A fragment: M = lane & 15
    const int hs   = lane >> 4;

    // Stage A: WAVES*16 s-rows, f32 -> f16, b128 global loads
    {
        const v4f* vbase4 = (const v4f*)(V + ((size_t)b * SS + sbase) * KK);
        for (int e = tid; e < WAVES * 16 * (KK / 4); e += WAVES * 32) {
            const int r  = e >> 7;        // / (KK/4)
            const int c4 = e & 127;
            const v4f x = vbase4[(size_t)r * (KK / 4) + c4];
            v4h y;
            y.x = (_Float16)x.x; y.y = (_Float16)x.y;
            y.z = (_Float16)x.z; y.w = (_Float16)x.w;
            *(v4h*)&ldsA[r >> 4][r & 15][c4 * 4] = y;
        }
    }

    float sacc[8];
#pragma unroll
    for (int r = 0; r < 8; ++r) sacc[r] = 0.f;

#if USE_TDM
    if (tid < 32) tdm_load_panel(panels, &ldsB[0][0]);   // prefetch panel 0
#endif

    for (int nt = 0; nt < NTILES; ++nt) {
#if USE_TDM
        const int cur = nt & 1;
        if (tid < 32) {
            if (nt + 1 < NTILES) {
                tdm_load_panel(panels + (size_t)(nt + 1) * PANEL_H,
                               &ldsB[1 - cur][0]);
                __builtin_amdgcn_s_wait_tensorcnt(1);   // panel nt complete
            } else {
                __builtin_amdgcn_s_wait_tensorcnt(0);
            }
        }
        __syncthreads();   // panel nt visible to all waves (also A on iter 0)
        const _Float16* bpan = &ldsB[cur][0];
#else
        __syncthreads();
        {   // synchronous staging fallback: pure b128 copies of packed panel
            const v4u* src = (const v4u*)(panels + (size_t)nt * PANEL_H);
            v4u* dst = (v4u*)&ldsB[0][0];
            for (int e = tid; e < PANEL_H / 8; e += WAVES * 32) dst[e] = src[e];
        }
        __syncthreads();
        const _Float16* bpan = &ldsB[0][0];
#endif

        v8f cacc = {};
#pragma unroll
        for (int ks = 0; ks < KK / 32; ++ks) {
            F16Frag a, bm;
            const int k0 = ks * 32;
            a.h[0]  = *(const v8h*)&ldsA[wave][row][k0 + 16 * hs];
            a.h[1]  = *(const v8h*)&ldsA[wave][row][k0 + 16 * hs + 8];
            bm.h[0] = *(const v8h*)&bpan[(ks * 32 + lane) * 16];
            bm.h[1] = *(const v8h*)&bpan[(ks * 32 + lane) * 16 + 8];
            cacc = __builtin_amdgcn_wmma_f32_16x16x32_f16(
                false, a.v, false, bm.v, (short)0, cacc, false, false);
        }
        // Epilogue: sacc[M] += v[h] * tanh(C[M,h] + proj_q[h]); N = lane&15
        const int   h   = nt * 16 + (lane & 15);
        const float pqv = pqh[h];
        const float vvh = vv[h];
#pragma unroll
        for (int r = 0; r < 8; ++r)
            sacc[r] += vvh * tanhf(cacc[r] + pqv);

        __syncthreads();   // all waves done with this panel buffer
    }

    // Reduce over N (16 lanes per half-wave); lanes 0/16 hold M=r / M=8+r
#pragma unroll
    for (int r = 0; r < 8; ++r) {
        sacc[r] += __shfl_xor(sacc[r], 1, 32);
        sacc[r] += __shfl_xor(sacc[r], 2, 32);
        sacc[r] += __shfl_xor(sacc[r], 4, 32);
        sacc[r] += __shfl_xor(sacc[r], 8, 32);
    }
    if ((lane & 15) == 0) {
        const int mb = (lane >> 4) * 8;
        float* sout = scores + ((size_t)head * BB + b) * SS + sbase + wave * 16 + mb;
#pragma unroll
        for (int r = 0; r < 8; ++r) sout[r] = sacc[r];
    }
}

// ---------------------------------------------------------------------------
// Kernel 3: sparsemax per (head,b) row via bisection + exact refinement.
// ---------------------------------------------------------------------------
__global__ void sparsemax_kernel(const float* __restrict__ scores,
                                 const unsigned char* __restrict__ mask_a,
                                 const unsigned char* __restrict__ mask_b,
                                 float* __restrict__ out) {
    const int b = blockIdx.x, head = blockIdx.y, tid = threadIdx.x;
    const unsigned char* mrow = (head ? mask_b : mask_a) + (size_t)b * SS;
    const float* srow = scores + ((size_t)head * BB + b) * SS;
    float* arow = out + (size_t)BB * HH + (size_t)head * BB * SS + (size_t)b * SS;

    __shared__ float red[256];
    __shared__ float red2[256];

    float z[16];
#pragma unroll
    for (int j = 0; j < 16; ++j) {
        const int idx = j * 256 + tid;
        z[j] = mrow[idx] ? srow[idx] : -1e9f;
    }

    float lmax = -3e38f;
#pragma unroll
    for (int j = 0; j < 16; ++j) lmax = fmaxf(lmax, z[j]);
    red[tid] = lmax;
    __syncthreads();
    for (int s = 128; s > 0; s >>= 1) {
        if (tid < s) red[tid] = fmaxf(red[tid], red[tid + s]);
        __syncthreads();
    }
    const float zmax = red[0];
    __syncthreads();

    float lo = zmax - 1.f, hi = zmax;
    for (int it = 0; it < 32; ++it) {
        const float mid = 0.5f * (lo + hi);
        float ls = 0.f;
#pragma unroll
        for (int j = 0; j < 16; ++j) ls += fmaxf(z[j] - mid, 0.f);
        red[tid] = ls;
        __syncthreads();
        for (int s = 128; s > 0; s >>= 1) {
            if (tid < s) red[tid] += red[tid + s];
            __syncthreads();
        }
        const float tot = red[0];
        __syncthreads();
        if (tot >= 1.f) lo = mid; else hi = mid;
    }

    float lc = 0.f, lsum = 0.f;
#pragma unroll
    for (int j = 0; j < 16; ++j)
        if (z[j] > lo) { lc += 1.f; lsum += z[j]; }
    red[tid] = lc; red2[tid] = lsum;
    __syncthreads();
    for (int s = 128; s > 0; s >>= 1) {
        if (tid < s) { red[tid] += red[tid + s]; red2[tid] += red2[tid + s]; }
        __syncthreads();
    }
    const float tau = (red2[0] - 1.f) / red[0];
    __syncthreads();

#pragma unroll
    for (int j = 0; j < 16; ++j)
        arow[j * 256 + tid] = fmaxf(z[j] - tau, 0.f);
}

// ---------------------------------------------------------------------------
// Kernel 4: context = alphas @ values, skipping alpha==0 rows (uniform branch)
// ---------------------------------------------------------------------------
__global__ void context_kernel(const float* __restrict__ va,
                               const float* __restrict__ vb,
                               const float* __restrict__ out,
                               float* __restrict__ ctx) {
    const int b = blockIdx.x, head = blockIdx.y, tid = threadIdx.x;
    const float* V = head ? vb : va;
    const float* arow = out + (size_t)BB * HH + (size_t)head * BB * SS + (size_t)b * SS;

    __shared__ float la[256];
    float acc0 = 0.f, acc1 = 0.f;
    for (int sc = 0; sc < SS / 256; ++sc) {
        const int s0 = sc * 256;
        __syncthreads();
        la[tid] = arow[s0 + tid];
        __syncthreads();
        for (int j = 0; j < 256; ++j) {
            const float a = la[j];
            if (a != 0.f) {
                const float* vr = V + ((size_t)b * SS + s0 + j) * KK;
                acc0 += a * vr[tid];
                acc1 += a * vr[tid + 256];
            }
        }
    }
    float* c = ctx + ((size_t)head * BB + b) * KK;
    c[tid] = acc0;
    c[tid + 256] = acc1;
}

// ---------------------------------------------------------------------------
// Kernel 5: gate merge -> att_vector in d_out[0 : B*H)
// ---------------------------------------------------------------------------
__global__ void gate_kernel(const float* __restrict__ query,
                            const float* __restrict__ ctx,
                            const float* __restrict__ Wg,
                            const float* __restrict__ bg,
                            const float* __restrict__ Wua,
                            const float* __restrict__ bua,
                            const float* __restrict__ Wub,
                            const float* __restrict__ bub,
                            float* __restrict__ out) {
    const int b = blockIdx.x, tid = threadIdx.x;
    const float* q  = query + (size_t)b * HH;
    const float* ca = ctx + (size_t)b * KK;
    const float* cb = ctx + (size_t)BB * KK + (size_t)b * KK;

    __shared__ float r0[HH], r1[HH];
    __shared__ float w01[2];

    float p0 = 0.f, p1 = 0.f;
    for (int i = tid; i < HH + 2 * KK; i += HH) {
        const float x = (i < HH) ? q[i] : (i < HH + KK ? ca[i - HH] : cb[i - HH - KK]);
        p0 += x * Wg[(size_t)i * 2 + 0];
        p1 += x * Wg[(size_t)i * 2 + 1];
    }
    r0[tid] = p0; r1[tid] = p1;
    __syncthreads();
    for (int s = HH / 2; s > 0; s >>= 1) {
        if (tid < s) { r0[tid] += r0[tid + s]; r1[tid] += r1[tid + s]; }
        __syncthreads();
    }
    if (tid == 0) {
        const float g0 = r0[0] + bg[0], g1 = r1[0] + bg[1];
        const float m  = fmaxf(g0, g1);
        const float e0 = expf(g0 - m), e1 = expf(g1 - m);
        const float inv = 1.f / (e0 + e1);
        w01[0] = e0 * inv; w01[1] = e1 * inv;
    }
    __syncthreads();

    float accA = 0.f, accB = 0.f;
    for (int i = 0; i < HH + KK; ++i) {
        const float xa = (i < HH) ? q[i] : ca[i - HH];
        const float xb = (i < HH) ? q[i] : cb[i - HH];
        accA += xa * Wua[(size_t)i * HH + tid];
        accB += xb * Wub[(size_t)i * HH + tid];
    }
    out[(size_t)b * HH + tid] =
        w01[0] * tanhf(accA + bua[tid]) + w01[1] * tanhf(accB + bub[tid]);
}

// ---------------------------------------------------------------------------
extern "C" void kernel_launch(void* const* d_in, const int* in_sizes, int n_in,
                              void* d_out, int out_size, void* d_ws, size_t ws_size,
                              hipStream_t stream) {
    const float* query    = (const float*)d_in[0];
    const float* values_a = (const float*)d_in[1];
    const float* values_b = (const float*)d_in[2];
    const unsigned char* mask_a = (const unsigned char*)d_in[3];
    const unsigned char* mask_b = (const unsigned char*)d_in[4];
    const float* Wk_a = (const float*)d_in[5];
    const float* Wq_a = (const float*)d_in[6];
    const float* v_a  = (const float*)d_in[7];
    const float* Wk_b = (const float*)d_in[8];
    const float* Wq_b = (const float*)d_in[9];
    const float* v_b  = (const float*)d_in[10];
    const float* Wg   = (const float*)d_in[11];
    const float* bg   = (const float*)d_in[12];
    const float* Wu_a = (const float*)d_in[13];
    const float* bu_a = (const float*)d_in[14];
    const float* Wu_b = (const float*)d_in[15];
    const float* bu_b = (const float*)d_in[16];

    float* out = (float*)d_out;

    // workspace layout
    float* ws     = (float*)d_ws;
    float* pq     = ws;                             // 2*B*H   floats
    float* scores = ws + 2 * BB * HH;               // 2*B*S   floats
    float* ctx    = ws + 2 * BB * HH + 2 * BB * SS; // 2*B*K   floats
    _Float16* wkp = (_Float16*)(ws + 2 * BB * HH + 2 * BB * SS + 2 * BB * KK);
                                                    // 2*K*H f16 (fragment-packed)

    {   // pack Wk -> f16 fragment order (L2-resident, ~1MB)
        dim3 g(2, NTILES);
        pack_wk_kernel<<<g, 256, 0, stream>>>(Wk_a, Wk_b, wkp);
    }
    {   // proj_q for both heads
        dim3 g(BB, 2);
        pq_kernel<<<g, HH, 0, stream>>>(query, Wq_a, Wq_b, pq);
    }
    {   // fused WMMA scores, TDM-fed double-buffered Wk panels
        dim3 g(SS / (WAVES * 16), BB, 2);
        scores_kernel<<<g, WAVES * 32, 0, stream>>>(values_a, values_b, wkp,
                                                    v_a, v_b, pq, scores);
    }
    {   // sparsemax -> alphas written into d_out
        dim3 g(BB, 2);
        sparsemax_kernel<<<g, 256, 0, stream>>>(scores, mask_a, mask_b, out);
    }
    {   // sparse context matvec
        dim3 g(BB, 2);
        context_kernel<<<g, 256, 0, stream>>>(values_a, values_b, out, ctx);
    }
    {   // gate merge -> att_vector in d_out[0:B*H)
        gate_kernel<<<BB, HH, 0, stream>>>(query, ctx, Wg, bg,
                                           Wu_a, bu_a, Wu_b, bu_b, out);
    }
}
